// LSTM_57088705299052
// MI455X (gfx1250) — matrix operand order repaired and checked by reading, not compiled, not run on hardware
//
#include <hip/hip_runtime.h>

// Fused 2-layer LSTM + linear head for MI455X (gfx1250, wave32, WMMA).
// One block = 16 batch rows, runs all 512 timesteps of BOTH layers internally.
// Weights live in VGPRs as pre-packed bf16 WMMA B-fragments; h/c state in LDS.
// v2: v_rcp_f32-based activations (no IEEE division on critical path),
//     double-buffered x staging -> 4 barriers per timestep.

typedef __attribute__((ext_vector_type(16))) __bf16 v16bf;
typedef __attribute__((ext_vector_type(8)))  float  v8f;

#define T_LEN    512
#define F_LEN    32
#define H_LEN    64
#define G_LEN    256   // 4*H
#define BTILE    16
#define NTHREADS 256   // 8 wave32 waves

__device__ __forceinline__ unsigned short f32_to_bf16(float f) {
  union { float f; unsigned int u; } x; x.f = f;
  unsigned int r = x.u + 0x7FFFu + ((x.u >> 16) & 1u);   // round-to-nearest-even
  return (unsigned short)(r >> 16);
}

// Fast activations: single v_exp_f32 + single v_rcp_f32, no v_div_scale chains.
__device__ __forceinline__ float fsigmoid(float v) {
  return __builtin_amdgcn_rcpf(1.0f + __expf(-v));
}
__device__ __forceinline__ float ftanhf(float v) {
  return 1.0f - 2.0f * __builtin_amdgcn_rcpf(__expf(2.0f * v) + 1.0f);
}

union ABfrag { v16bf v; unsigned int u[8]; };

// A-fragment (16x32 bf16, MxK) from LDS row-major buffer.
// ISA layout: lane(0..15)->M, lane>=16 selects upper K-half; VGPR v: grp=v>>2
// (K base 0/16), pos=v&3 -> K pair {grp*16 + khalf*8 + 2*pos, +1}.
__device__ __forceinline__ v16bf load_a16x32(const unsigned short* base, int rowstride,
                                             int kbase, int lane) {
  ABfrag r;
  int M = lane & 15;
  int khalf = lane >> 4;
  #pragma unroll
  for (int v = 0; v < 8; ++v) {
    int grp = v >> 2, pos = v & 3;
    int K = kbase + grp * 16 + khalf * 8 + pos * 2;
    r.u[v] = *(const unsigned int*)(base + M * rowstride + K);   // ds_load
  }
  return r.v;
}

// B-fragment (32x16 bf16, KxN) built from global fp32 weight W[4H][Kdim].
// gates = h @ W^T  =>  B[k][n] = W[nc+n][k].  ISA layout: lane(0..15)->N, lane>=16 selects upper K-half; K order as A.
__device__ __forceinline__ v16bf load_b32x16(const float* __restrict__ W, int Kdim,
                                             int nc, int kbase, int lane) {
  ABfrag r;
  int N = nc + (lane & 15);
  int khalf = lane >> 4;
  #pragma unroll
  for (int v = 0; v < 8; ++v) {
    int K = kbase + (v >> 2) * 16 + khalf * 8 + (v & 3) * 2;
    unsigned int lo = f32_to_bf16(W[(size_t)N * Kdim + K    ]);
    unsigned int hi = f32_to_bf16(W[(size_t)N * Kdim + K + 1]);
    r.u[v] = lo | (hi << 16);
  }
  return r.v;
}

// C/D fragment scatter to LDS gates[16][256].
// ISA layout: VGPR r, lane: M = r + 8*(lane>>4), N = lane&15.
__device__ __forceinline__ void store_cfrag(float* g, int colbase, int lane, v8f c) {
  int N  = lane & 15;
  int mh = (lane >> 4) * 8;
  #pragma unroll
  for (int r = 0; r < 8; ++r) g[(mh + r) * G_LEN + colbase + N] = c[r];
}

#define WMMA_BF16(a, b, c) \
  __builtin_amdgcn_wmma_f32_16x16x32_bf16(false, (a), false, (b), (short)0, (c), false, false)

__global__ __launch_bounds__(NTHREADS) void lstm2_fused_kernel(
    const float* __restrict__ x,
    const float* __restrict__ W_ih0, const float* __restrict__ W_hh0,
    const float* __restrict__ b_ih0, const float* __restrict__ b_hh0,
    const float* __restrict__ W_ih1, const float* __restrict__ W_hh1,
    const float* __restrict__ b_ih1, const float* __restrict__ b_hh1,
    const float* __restrict__ fc_w,  const float* __restrict__ fc_b,
    float* __restrict__ out)
{
  __shared__ __align__(16) float          gates[BTILE * G_LEN];      // 16 KB
  __shared__ __align__(16) float          c0s  [BTILE * H_LEN];      // fp32 cell, layer 0
  __shared__ __align__(16) float          c1s  [BTILE * H_LEN];      // fp32 cell, layer 1
  __shared__ __align__(16) float          hf32 [BTILE * H_LEN];      // fp32 h2 (head)
  __shared__ __align__(16) unsigned short h0s  [BTILE * H_LEN];      // bf16 h, layer 0
  __shared__ __align__(16) unsigned short h1s  [BTILE * H_LEN];      // bf16 h, layer 1
  __shared__ __align__(16) unsigned short xin  [2][BTILE * F_LEN];   // bf16 x_t, 2-deep

  const int tid     = threadIdx.x;
  const int lane    = tid & 31;
  const int wave    = tid >> 5;
  const int b0      = blockIdx.x * BTILE;
  const int colbase = wave * 32;          // each wave owns gate columns [32w, 32w+32)

  for (int e = tid; e < BTILE * H_LEN; e += NTHREADS) {
    c0s[e] = 0.f; c1s[e] = 0.f; hf32[e] = 0.f; h0s[e] = 0; h1s[e] = 0;
  }

  // ---- Pin all weights for this wave's 2 N-tiles in VGPRs (one-time) ----
  v16bf Bih0[2], Bhh0[2][2], Bih1[2][2], Bhh1[2][2];
  v8f   bias0[2], bias1[2];
  #pragma unroll
  for (int tile = 0; tile < 2; ++tile) {
    int nc = colbase + tile * 16;
    Bih0[tile] = load_b32x16(W_ih0, F_LEN, nc, 0, lane);
    #pragma unroll
    for (int kc = 0; kc < 2; ++kc) {
      Bhh0[tile][kc] = load_b32x16(W_hh0, H_LEN, nc, kc * 32, lane);
      Bih1[tile][kc] = load_b32x16(W_ih1, H_LEN, nc, kc * 32, lane);
      Bhh1[tile][kc] = load_b32x16(W_hh1, H_LEN, nc, kc * 32, lane);
    }
    float bs0 = b_ih0[nc + (lane & 15)] + b_hh0[nc + (lane & 15)];
    float bs1 = b_ih1[nc + (lane & 15)] + b_hh1[nc + (lane & 15)];
    #pragma unroll
    for (int r = 0; r < 8; ++r) { bias0[tile][r] = bs0; bias1[tile][r] = bs1; }
  }

  // ---- Prologue: stage x_0 ----
  #pragma unroll
  for (int i = 0; i < 2; ++i) {
    int e = tid + i * NTHREADS;
    int row = e >> 5, col = e & 31;
    xin[0][e] = f32_to_bf16(x[((size_t)(b0 + row) * T_LEN + 0) * F_LEN + col]);
  }
  __syncthreads();

  // ---- Sequential recurrence over T, both layers fused (4 barriers/step) ----
  for (int t = 0; t < T_LEN; ++t) {
    const int cur = t & 1, nxt = cur ^ 1;

    // ---- Layer 0: gates = bias + x_t@Wih0^T + h0@Whh0^T ----
    {
      v16bf Ax  = load_a16x32(xin[cur], F_LEN, 0,  lane);
      v16bf Ah0 = load_a16x32(h0s,      H_LEN, 0,  lane);
      v16bf Ah1 = load_a16x32(h0s,      H_LEN, 32, lane);
      #pragma unroll
      for (int tile = 0; tile < 2; ++tile) {
        v8f c = bias0[tile];
        c = WMMA_BF16(Ax,  Bih0[tile],    c);
        c = WMMA_BF16(Ah0, Bhh0[tile][0], c);
        c = WMMA_BF16(Ah1, Bhh0[tile][1], c);
        store_cfrag(gates, colbase + tile * 16, lane, c);
      }
    }
    __syncthreads();

    // Elementwise layer 0 -> c0, h0; also stage x_{t+1} into the other buffer.
    #pragma unroll
    for (int i = 0; i < 4; ++i) {
      int e = tid + i * NTHREADS;
      int row = e >> 6, j = e & 63;
      const float* g = gates + row * G_LEN + j;
      float ii = fsigmoid(g[0]);
      float ff = fsigmoid(g[64]);
      float gg = ftanhf (g[128]);
      float oo = fsigmoid(g[192]);
      float cn = ff * c0s[e] + ii * gg;
      c0s[e] = cn;
      h0s[e] = f32_to_bf16(oo * ftanhf(cn));
    }
    if (t + 1 < T_LEN) {
      #pragma unroll
      for (int i = 0; i < 2; ++i) {
        int e = tid + i * NTHREADS;
        int row = e >> 5, col = e & 31;
        const float* xp = x + ((size_t)(b0 + row) * T_LEN + (t + 1)) * F_LEN + col;
        xin[nxt][e] = f32_to_bf16(*xp);
        if (t + 2 < T_LEN) __builtin_prefetch(xp + F_LEN, 0, 3);  // global_prefetch_b8
      }
    }
    __syncthreads();

    // ---- Layer 1: gates = bias + h0@Wih1^T + h1@Whh1^T ----
    {
      v16bf Ai0 = load_a16x32(h0s, H_LEN, 0,  lane);
      v16bf Ai1 = load_a16x32(h0s, H_LEN, 32, lane);
      v16bf Ar0 = load_a16x32(h1s, H_LEN, 0,  lane);
      v16bf Ar1 = load_a16x32(h1s, H_LEN, 32, lane);
      #pragma unroll
      for (int tile = 0; tile < 2; ++tile) {
        v8f c = bias1[tile];
        c = WMMA_BF16(Ai0, Bih1[tile][0], c);
        c = WMMA_BF16(Ai1, Bih1[tile][1], c);
        c = WMMA_BF16(Ar0, Bhh1[tile][0], c);
        c = WMMA_BF16(Ar1, Bhh1[tile][1], c);
        store_cfrag(gates, colbase + tile * 16, lane, c);
      }
    }
    __syncthreads();

    // Elementwise layer 1 -> c1, h1 (bf16) and fp32 copy for the head
    #pragma unroll
    for (int i = 0; i < 4; ++i) {
      int e = tid + i * NTHREADS;
      int row = e >> 6, j = e & 63;
      const float* g = gates + row * G_LEN + j;
      float ii = fsigmoid(g[0]);
      float ff = fsigmoid(g[64]);
      float gg = ftanhf (g[128]);
      float oo = fsigmoid(g[192]);
      float cn = ff * c1s[e] + ii * gg;
      float hn = oo * ftanhf(cn);
      c1s[e] = cn;
      h1s[e] = f32_to_bf16(hn);
      hf32[e] = hn;
    }
    __syncthreads();
  }

  // ---- Linear head: out[b] = h2_last[b,:] . fc_w + fc_b ----
  if (tid < BTILE) {
    float acc = fc_b[0];
    #pragma unroll
    for (int k = 0; k < H_LEN; ++k) acc += hf32[tid * H_LEN + k] * fc_w[k];
    out[b0 + tid] = acc;
  }
}

extern "C" void kernel_launch(void* const* d_in, const int* in_sizes, int n_in,
                              void* d_out, int out_size, void* d_ws, size_t ws_size,
                              hipStream_t stream) {
  const float* x     = (const float*)d_in[0];
  const float* W_ih0 = (const float*)d_in[1];
  const float* W_hh0 = (const float*)d_in[2];
  const float* b_ih0 = (const float*)d_in[3];
  const float* b_hh0 = (const float*)d_in[4];
  const float* W_ih1 = (const float*)d_in[5];
  const float* W_hh1 = (const float*)d_in[6];
  const float* b_ih1 = (const float*)d_in[7];
  const float* b_hh1 = (const float*)d_in[8];
  const float* fc_w  = (const float*)d_in[9];
  const float* fc_b  = (const float*)d_in[10];
  float* out = (float*)d_out;

  int B    = in_sizes[0] / (T_LEN * F_LEN);   // 1024
  int grid = B / BTILE;                        // 64 blocks x 256 threads

  lstm2_fused_kernel<<<grid, NTHREADS, 0, stream>>>(
      x, W_ih0, W_hh0, b_ih0, b_hh0, W_ih1, W_hh1, b_ih1, b_hh1, fc_w, fc_b, out);

  (void)n_in; (void)out_size; (void)d_ws; (void)ws_size;
}
